// SimpleActor_3994319585781
// MI455X (gfx1250) — matrix operand order repaired and checked by reading, not compiled, not run on hardware
//
#include <hip/hip_runtime.h>

#define B_  32
#define T_  128
#define Z_  100
#define S_  6
#define H_  64
#define G3  192     // 3*H
#define F1_ 512
#define F2_ 128
#define ZB  3200    // Z_*B_
#define KP  96      // padded GRU K: [x(6) pad->32 | h(64)]

typedef __attribute__((ext_vector_type(16))) __bf16 v16bf;
typedef __attribute__((ext_vector_type(8)))  __bf16 v8bf;
typedef __attribute__((ext_vector_type(8)))  float  v8f;

// Load a 16x16x32 WMMA operand for one lane: two contiguous 16B chunks.
static __device__ inline v16bf ldpair(const __bf16* p0, const __bf16* p1) {
  union { v16bf v; v8bf h[2]; } u;
  u.h[0] = *(const v8bf*)p0;
  u.h[1] = *(const v8bf*)p1;
  return u.v;
}

static __device__ inline v8f bcast8(float s) {
  v8f c;
#pragma unroll
  for (int e = 0; e < 8; ++e) c[e] = s;
  return c;
}

// Fast activations on v_exp_f32 + v_rcp_f32 (avoid IEEE div expansion).
static __device__ inline float fast_sigmoid(float v) {
  return __builtin_amdgcn_rcpf(1.0f + __expf(-v));
}
static __device__ inline float fast_tanh(float v) {
  float ex = __expf(-2.0f * v);                 // e^{-2v}
  return (1.0f - ex) * __builtin_amdgcn_rcpf(1.0f + ex);
}

// ---------------- kernel 1: fused GRU scan (16 sequences per wave) ----------------
// One wave per workgroup: 200 blocks spread the latency-bound scan across WGPs.
__global__ __launch_bounds__(32) void gru_kernel(
    const float* __restrict__ x, const float* __restrict__ Wih,
    const float* __restrict__ Whh, const float* __restrict__ bih,
    const float* __restrict__ bhh, __bf16* __restrict__ sx) {
  // Combined weights, B-tile friendly layout [g][k]: k<6 Wih, 6..31 zero, 32..95 Whh
  __shared__ __bf16 Wc[G3 * KP];      // 36,864 B
  __shared__ __bf16 hw[16 * KP];      // [seq][k] staging: k0..5 x, 6..31 zero, 32..95 h

  const int lane = threadIdx.x & 31;
  for (int i = lane; i < G3 * KP; i += 32) {
    int g = i / KP, k = i - g * KP;
    float v = (k < S_) ? Wih[g * S_ + k] : ((k < 32) ? 0.0f : Whh[g * H_ + (k - 32)]);
    Wc[i] = (__bf16)v;
  }
  __syncthreads();

  const int gcol = lane & 15, hi = lane >> 4;
  float bsum[8], bin[4], bhn[4];
#pragma unroll
  for (int nt = 0; nt < 8; ++nt) bsum[nt] = bih[nt * 16 + gcol] + bhh[nt * 16 + gcol];
#pragma unroll
  for (int j = 0; j < 4; ++j) {
    bin[j] = bih[128 + j * 16 + gcol];
    bhn[j] = bhh[128 + j * 16 + gcol];
  }

  for (int tile = blockIdx.x; tile < ZB / 16; tile += gridDim.x) {
    const int n0 = tile * 16;
    // zero the whole staging buffer (x-pad region + h(0)=0)
#pragma unroll
    for (int i = 0; i < 6; ++i) {
      v8bf zv = {};
      *(v8bf*)(hw + (i * 32 + lane) * 8) = zv;
    }
    v8f hD[4];
#pragma unroll
    for (int j = 0; j < 4; ++j) hD[j] = bcast8(0.0f);

    for (int t = 0; t < T_; ++t) {
      // stage x_t (16 seqs x 6) into cols 0..5 (3 elements per lane)
#pragma unroll
      for (int q = 0; q < 3; ++q) {
        int idx = lane * 3 + q;
        int j = idx / S_, s = idx - j * S_;
        int n = n0 + j, zz = n >> 5, bb = n & 31;
        hw[j * KP + s] = (__bf16)x[((size_t)(bb * T_ + t) * Z_ + zz) * S_ + s];
      }
      // prefetch next timestep's x gather (global_prefetch_b8)
      if (t + 1 < T_) {
        int idx = lane * 3;
        int j = idx / S_, s = idx - j * S_;
        int n = n0 + j, zz = n >> 5, bb = n & 31;
        __builtin_prefetch(&x[((size_t)(bb * T_ + (t + 1)) * Z_ + zz) * S_ + s], 0, 1);
      }
      // A tiles: [x|h] row-major, contiguous k = 16*hi + s
      v16bf A[3];
#pragma unroll
      for (int kc = 0; kc < 3; ++kc) {
        const __bf16* p = hw + (lane & 15) * KP + kc * 32 + hi * 16;
        A[kc] = ldpair(p, p + 8);
      }
      // r,z gates: combined input+hidden accumulation, bias pre-added
      v8f acc[8];
#pragma unroll
      for (int nt = 0; nt < 8; ++nt) {
        v8f c = bcast8(bsum[nt]);
#pragma unroll
        for (int kc = 0; kc < 3; ++kc) {
          const __bf16* p = Wc + (nt * 16 + gcol) * KP + kc * 32 + hi * 16;
          v16bf Bm = ldpair(p, p + 8);
          c = __builtin_amdgcn_wmma_f32_16x16x32_bf16(false, A[kc], false, Bm,
                                                      (short)0, c, false, false);
        }
        acc[nt] = c;
      }
      // n gate: input part and hidden part kept separate for r * gh_n
#pragma unroll
      for (int j = 0; j < 4; ++j) {
        const __bf16* pb = Wc + ((8 + j) * 16 + gcol) * KP;
        v16bf B0 = ldpair(pb + hi * 16, pb + hi * 16 + 8);
        v8f ci = bcast8(bin[j]);
        ci = __builtin_amdgcn_wmma_f32_16x16x32_bf16(false, A[0], false, B0,
                                                     (short)0, ci, false, false);
        v8f ch = bcast8(bhn[j]);
#pragma unroll
        for (int kc = 1; kc < 3; ++kc) {
          const __bf16* p = pb + kc * 32 + hi * 16;
          v16bf Bm = ldpair(p, p + 8);
          ch = __builtin_amdgcn_wmma_f32_16x16x32_bf16(false, A[kc], false, Bm,
                                                       (short)0, ch, false, false);
        }
#pragma unroll
        for (int e = 0; e < 8; ++e) {
          float r  = fast_sigmoid(acc[j][e]);
          float zg = fast_sigmoid(acc[4 + j][e]);
          float nn = fast_tanh(ci[e] + r * ch[e]);
          hD[j][e] = nn + zg * (hD[j][e] - nn);   // (1-z)*n + z*h as one FMA chain
        }
        // write h back to staging (D layout -> row major bf16)
#pragma unroll
        for (int e = 0; e < 8; ++e) {
          int seq = e + hi * 8;
          hw[seq * KP + 32 + j * 16 + gcol] = (__bf16)hD[j][e];
        }
      }
      // stream 16x64 h tile to sx (coalesced 16B stores)
#pragma unroll
      for (int i = 0; i < 4; ++i) {
        int o = i * 32 + lane;
        int row = o >> 3, cb = (o & 7) * 8;
        v8bf v = *(const v8bf*)(hw + row * KP + 32 + cb);
        *(v8bf*)(sx + ((size_t)(n0 + row) * T_ + t) * H_ + cb) = v;
      }
    }
  }
}

// ---------------- kernel 2: MLP 64->512->128->1 over 409600 rows ----------------
// All weights LDS-resident (256 KB/workgroup: CDNA5 allows up to 320 KB).
__global__ __launch_bounds__(128) void mlp_kernel(
    const __bf16* __restrict__ sx, const float* __restrict__ w1,
    const float* __restrict__ b1, const float* __restrict__ w2,
    const float* __restrict__ b2, const float* __restrict__ w3,
    const float* __restrict__ b3, float* __restrict__ out) {
  __shared__ __bf16 w1s[F1_ * H_];        //  64 KB  [g][k]
  __shared__ __bf16 w2s[F2_ * F1_];       // 128 KB  [g][k]
  __shared__ __bf16 a1s[4][16 * F1_];     //  64 KB  per-wave D->A relayout bounce
  const int tid = threadIdx.x, lane = tid & 31, wave = tid >> 5;
  const int gcol = lane & 15, hi = lane >> 4;

  // cooperative fp32 -> bf16 weight load (float4-vectorized, one-time per block)
  for (int i = tid * 4; i < F1_ * H_; i += 128 * 4) {
    float4 v = *(const float4*)(w1 + i);
    w1s[i + 0] = (__bf16)v.x; w1s[i + 1] = (__bf16)v.y;
    w1s[i + 2] = (__bf16)v.z; w1s[i + 3] = (__bf16)v.w;
  }
  for (int i = tid * 4; i < F2_ * F1_; i += 128 * 4) {
    float4 v = *(const float4*)(w2 + i);
    w2s[i + 0] = (__bf16)v.x; w2s[i + 1] = (__bf16)v.y;
    w2s[i + 2] = (__bf16)v.z; w2s[i + 3] = (__bf16)v.w;
  }
  __syncthreads();

  float b1l[32], b2l[8], w3l[8];
#pragma unroll
  for (int nt = 0; nt < 32; ++nt) b1l[nt] = b1[nt * 16 + gcol];
#pragma unroll
  for (int nt = 0; nt < 8; ++nt) { b2l[nt] = b2[nt * 16 + gcol]; w3l[nt] = w3[nt * 16 + gcol]; }
  const float b3s = b3[0];
  __bf16* as = a1s[wave];

  const int NT = (ZB * T_) / 16;          // 25600 row tiles
  for (int mt = blockIdx.x * 4 + wave; mt < NT; mt += gridDim.x * 4) {
    const size_t rowbase = (size_t)mt * 16;
    // prefetch next M-tile's activation rows
    {
      int mtn = mt + gridDim.x * 4;
      if (mtn < NT)
        __builtin_prefetch(sx + ((size_t)mtn * 16 + (lane & 15)) * H_, 0, 1);
    }
    // A (16x64) straight from sx (row major)
    const __bf16* pA = sx + (rowbase + (lane & 15)) * H_ + hi * 16;
    v16bf A0 = ldpair(pA, pA + 8);
    v16bf A1 = ldpair(pA + 32, pA + 40);

    // layer 1: 32 N-tiles x 2 K-chunks, B from LDS
#pragma unroll
    for (int nt = 0; nt < 32; ++nt) {
      const __bf16* p = w1s + (nt * 16 + gcol) * H_ + hi * 16;
      v16bf B0 = ldpair(p, p + 8);
      v16bf B1 = ldpair(p + 32, p + 40);
      v8f c = bcast8(b1l[nt]);
      c = __builtin_amdgcn_wmma_f32_16x16x32_bf16(false, A0, false, B0, (short)0, c, false, false);
      c = __builtin_amdgcn_wmma_f32_16x16x32_bf16(false, A1, false, B1, (short)0, c, false, false);
#pragma unroll
      for (int e = 0; e < 8; ++e) {
        float v = c[e] > 0.0f ? c[e] : 0.0f;                 // ReLU
        as[(e + hi * 8) * F1_ + nt * 16 + gcol] = (__bf16)v; // D->A relayout via LDS
      }
    }
    // layer 2: 8 N-tiles x 16 K-chunks, B from LDS
    v8f acc2[8];
#pragma unroll
    for (int nt = 0; nt < 8; ++nt) acc2[nt] = bcast8(b2l[nt]);
    for (int kc = 0; kc < 16; ++kc) {
      const __bf16* pa = as + (lane & 15) * F1_ + kc * 32 + hi * 16;
      v16bf A2 = ldpair(pa, pa + 8);
#pragma unroll
      for (int nt = 0; nt < 8; ++nt) {
        const __bf16* p = w2s + (nt * 16 + gcol) * F1_ + kc * 32 + hi * 16;
        v16bf Bm = ldpair(p, p + 8);
        acc2[nt] = __builtin_amdgcn_wmma_f32_16x16x32_bf16(false, A2, false, Bm,
                                                           (short)0, acc2[nt], false, false);
      }
    }
    // layer 3: ReLU + dot with w3, butterfly reduce over each 16-lane half
    float ps[8];
#pragma unroll
    for (int e = 0; e < 8; ++e) ps[e] = 0.0f;
#pragma unroll
    for (int nt = 0; nt < 8; ++nt)
#pragma unroll
      for (int e = 0; e < 8; ++e) {
        float v = acc2[nt][e] > 0.0f ? acc2[nt][e] : 0.0f;
        ps[e] += v * w3l[nt];
      }
#pragma unroll
    for (int e = 0; e < 8; ++e) {
      ps[e] += __shfl_xor(ps[e], 1, 32);
      ps[e] += __shfl_xor(ps[e], 2, 32);
      ps[e] += __shfl_xor(ps[e], 4, 32);
      ps[e] += __shfl_xor(ps[e], 8, 32);
    }
    if (gcol == 0) {
#pragma unroll
      for (int e = 0; e < 8; ++e) {
        int m = (int)rowbase + e + hi * 8;
        int n = m >> 7, t = m & 127;     // T = 128
        int zz = n >> 5, bb = n & 31;    // n = z*B + b, B = 32
        out[(size_t)bb * (T_ * Z_) + t * Z_ + zz] = ps[e] + b3s;
      }
    }
  }
}

extern "C" void kernel_launch(void* const* d_in, const int* in_sizes, int n_in,
                              void* d_out, int out_size, void* d_ws, size_t ws_size,
                              hipStream_t stream) {
  const float* x   = (const float*)d_in[0];
  const float* Wih = (const float*)d_in[1];
  const float* Whh = (const float*)d_in[2];
  const float* bih = (const float*)d_in[3];
  const float* bhh = (const float*)d_in[4];
  const float* w1  = (const float*)d_in[5];
  const float* b1  = (const float*)d_in[6];
  const float* w2  = (const float*)d_in[7];
  const float* b2  = (const float*)d_in[8];
  const float* w3  = (const float*)d_in[9];
  const float* b3  = (const float*)d_in[10];
  float* out = (float*)d_out;

  // ws layout: sx (ZB*T*H bf16, 52.4 MB)
  __bf16* sx = (__bf16*)d_ws;

  gru_kernel<<<200, 32, 0, stream>>>(x, Wih, Whh, bih, bhh, sx);
  mlp_kernel<<<320, 128, 0, stream>>>(sx, w1, b1, w2, b2, w3, b3, out);
}
